// InterleavedHeadAttention_34591666601984
// MI455X (gfx1250) — matrix operand order repaired and checked by reading, not compiled, not run on hardware
//
#include <hip/hip_runtime.h>
#include <hip/hip_bf16.h>

// ---------------------------------------------------------------------------
// InterleavedHeadAttention for MI455X (gfx1250, wave32, WMMA bf16).
// alpha-mixing folded into Wq/Wk/Wv; collapse folded into Wo.
// Pipeline: prep -> QKV GEMM (wmma bf16) -> causal flash attention (wmma bf16)
//           -> output GEMM (wmma bf16, f32 out).
// GLOBAL_LOAD_ASYNC_TO_LDS_B128 staging (ASYNCcnt) with triple buffering;
// falls back to the proven sync path if builtins are absent.
// ---------------------------------------------------------------------------

typedef __bf16 bf16;
typedef __attribute__((ext_vector_type(16))) __bf16 v16bf;
typedef __attribute__((ext_vector_type(8)))  __bf16 v8bf;
typedef __attribute__((ext_vector_type(8)))  float  v8f;
typedef __attribute__((ext_vector_type(4)))  int    v4i;

#define WMMA_BF16(a, b, c) \
    __builtin_amdgcn_wmma_f32_16x16x32_bf16(false, (a), false, (b), (short)0, (c), false, false)

#if defined(__AMDGCN__) && __has_builtin(__builtin_amdgcn_global_load_async_to_lds_b128) && \
    __has_builtin(__builtin_amdgcn_s_wait_asynccnt)
#define USE_ASYNC_LDS 1
#else
#define USE_ASYNC_LDS 0
#endif

#if USE_ASYNC_LDS
typedef __attribute__((address_space(1))) v4i* gv4i_p;
typedef __attribute__((address_space(3))) v4i* lv4i_p;
__device__ __forceinline__ void async_cp16(const bf16* g, bf16* l) {
    __builtin_amdgcn_global_load_async_to_lds_b128((gv4i_p)g, (lv4i_p)l, 0, 0);
}
#endif

__device__ __forceinline__ v16bf cat8(v8bf lo, v8bf hi) {
    return __builtin_shufflevector(lo, hi, 0,1,2,3,4,5,6,7,8,9,10,11,12,13,14,15);
}
__device__ __forceinline__ v8f vzero8() {
    v8f v;
#pragma unroll
    for (int i = 0; i < 8; ++i) v[i] = 0.0f;
    return v;
}

static constexpr int   SV     = 2048;    // virtual sequence = S*P
static constexpr float SCALE  = 0.125f;  // 1/sqrt(64)
static constexpr float NEGBIG = -3.0e38f;

// ---------------------------------------------------------------------------
// prep: cast hidden_states to bf16
__global__ void prep_x(const float* __restrict__ x, bf16* __restrict__ xb) {
    int i = (blockIdx.x * 256 + threadIdx.x) * 4;
    float4 f = *(const float4*)(x + i);
    xb[i + 0] = (bf16)f.x; xb[i + 1] = (bf16)f.y;
    xb[i + 2] = (bf16)f.z; xb[i + 3] = (bf16)f.w;
}

// prep: W_effT[e][n] = sum_m alpha[m, h, p] * W[m*64+d, e],  n = (h*2+p)*64+d.
// Stored K-major (1024 x 2048) so the GEMM reads it as B^T directly.
__global__ void prep_weff(const float* __restrict__ W, const float* __restrict__ alpha,
                          bf16* __restrict__ WT) {
    int e  = blockIdx.x * 256 + threadIdx.x;   // 0..1023 (K index)
    int n0 = blockIdx.y * 16;                  // 0..2047 step 16 (N index)
    int vh = n0 >> 6;                          // uniform within block
    int d0 = n0 & 63;
    float acc[16];
#pragma unroll
    for (int i = 0; i < 16; ++i) acc[i] = 0.0f;
    for (int m = 0; m < 16; ++m) {
        float a = alpha[m * 32 + vh];
#pragma unroll
        for (int i = 0; i < 16; ++i)
            acc[i] += a * W[(size_t)(m * 64 + d0 + i) * 1024 + e];
    }
#pragma unroll
    for (int i = 0; i < 16; ++i)
        WT[(size_t)e * 2048 + n0 + i] = (bf16)acc[i];
}

// prep: Wo_effT[k][f] = sum_h Wo[f, h*64+dd] * collapse[h, o],  k = o*64+dd.
// Stored K-major (2048 x 1024).
__global__ void prep_woeff(const float* __restrict__ Wo, const float* __restrict__ col,
                           bf16* __restrict__ WT) {
    int k  = blockIdx.x * 256 + threadIdx.x;   // 0..2047
    int o  = k >> 6, dd = k & 63;
    int f0 = blockIdx.y * 16;
    float acc[16];
#pragma unroll
    for (int i = 0; i < 16; ++i) acc[i] = 0.0f;
    for (int h = 0; h < 16; ++h) {
        float c = col[h * 32 + o];
#pragma unroll
        for (int i = 0; i < 16; ++i)
            acc[i] += c * Wo[(size_t)(f0 + i) * 1024 + h * 64 + dd];
    }
#pragma unroll
    for (int i = 0; i < 16; ++i)
        WT[(size_t)k * 1024 + f0 + i] = (bf16)acc[i];
}

// ---------------------------------------------------------------------------
// GEMM: C(MxN) = A(MxK) * BT(KxN), bf16 in, f32 accumulate.
// 256 threads = 8 waves; block tile 128x128; wave tile 32x64; K-step 32.
// Async path: triple-buffered LDS tiles, one barrier per K-step.
// mode: 0 -> Q[b,h,vpos,d] bf16   1 -> K^T[b,h,d,vpos] bf16
//       2 -> V[b,h,vpos,d] bf16   3 -> plain f32 row-major (final output)
__global__ void __launch_bounds__(256) gemm_wmma(
    const bf16* __restrict__ A, const bf16* __restrict__ BT,
    void* __restrict__ Cout, int M, int N, int K, int mode)
{
#if USE_ASYNC_LDS
    __shared__ __align__(16) bf16 As[3][128 * 40];   // 128 rows x 32 k (pad 40)
    __shared__ __align__(16) bf16 Bs[3][32 * 136];   // 32 k x 128 n (pad 136)
#else
    __shared__ __align__(16) bf16 As[1][128 * 40];
    __shared__ __align__(16) bf16 Bs[1][32 * 136];
#endif

    const int tid  = threadIdx.x;
    const int wid  = tid >> 5, lane = tid & 31;
    const int lm   = lane & 15, half = lane >> 4;
    const int bm   = blockIdx.y * 128, bn = blockIdx.x * 128;
    const int wm   = (wid >> 1) * 32, wn = (wid & 1) * 64;

    // cooperative-load indices
    const int ar = tid >> 1, acs = (tid & 1) * 16;
    const int br = tid >> 3, bcs = (tid & 7) * 16;
    const bf16* gA = A + (size_t)(bm + ar) * K + acs;   // + k0
    const bf16* gB = BT + (size_t)br * N + bn + bcs;    // + k0*N

    v8f acc[2][4];
#pragma unroll
    for (int mt = 0; mt < 2; ++mt)
#pragma unroll
        for (int nt = 0; nt < 4; ++nt) acc[mt][nt] = vzero8();

#if USE_ASYNC_LDS
    auto issue = [&](int k0, int buf) {
        const bf16* ga = gA + k0;
        async_cp16(ga,     &As[buf][ar * 40 + acs]);
        async_cp16(ga + 8, &As[buf][ar * 40 + acs + 8]);
        const bf16* gb = gB + (size_t)k0 * N;
        async_cp16(gb,     &Bs[buf][br * 136 + bcs]);
        async_cp16(gb + 8, &Bs[buf][br * 136 + bcs + 8]);
    };
    issue(0, 0);
#endif

    const int nsteps = K >> 5;
    for (int i = 0; i < nsteps; ++i) {
#if USE_ASYNC_LDS
        const int buf = i % 3;
        if (i + 1 < nsteps) {
            issue(i * 32 + 32, (i + 1) % 3);
            __builtin_amdgcn_s_wait_asynccnt(4);   // my tile-i copies done
        } else {
            __builtin_amdgcn_s_wait_asynccnt(0);
        }
        __syncthreads();                           // everyone's tile-i done
#else
        const int buf = 0;
        const int k0 = i * 32;
        __syncthreads();
        {
            const bf16* g = gA + k0;
            *(uint4*)&As[0][ar * 40 + acs]     = *(const uint4*)g;
            *(uint4*)&As[0][ar * 40 + acs + 8] = *(const uint4*)(g + 8);
            const bf16* gb = gB + (size_t)k0 * N;
            *(uint4*)&Bs[0][br * 136 + bcs]     = *(const uint4*)gb;
            *(uint4*)&Bs[0][br * 136 + bcs + 8] = *(const uint4*)(gb + 8);
            if (k0 + 32 < K) {
                __builtin_prefetch(g + 32, 0, 3);
                __builtin_prefetch(gb + (size_t)32 * N, 0, 3);
            }
        }
        __syncthreads();
#endif

        v16bf af[2], bfm[4];
#pragma unroll
        for (int mt = 0; mt < 2; ++mt) {   // A frag: lane=row, K pairs in vgprs
            const bf16* p = &As[buf][(wm + mt * 16 + lm) * 40 + half * 8];
            af[mt] = cat8(*(const v8bf*)p, *(const v8bf*)(p + 16));
        }
        const int kb = half * 8;           // B frag: lane=N col, K in table order
#pragma unroll
        for (int nt = 0; nt < 4; ++nt) {
            const bf16* p = &Bs[buf][kb * 136 + wn + nt * 16 + lm];
#pragma unroll
            for (int s = 0; s < 8; ++s) {
                bfm[nt][s]     = p[s * 136];
                bfm[nt][s + 8] = p[(s + 16) * 136];
            }
        }
#pragma unroll
        for (int mt = 0; mt < 2; ++mt)
#pragma unroll
            for (int nt = 0; nt < 4; ++nt)
                acc[mt][nt] = WMMA_BF16(af[mt], bfm[nt], acc[mt][nt]);
    }

    // Epilogue. C frag: lane holds [m=(half*8+j), n=lm].
#pragma unroll
    for (int mt = 0; mt < 2; ++mt)
#pragma unroll
        for (int nt = 0; nt < 4; ++nt)
#pragma unroll
            for (int j = 0; j < 8; ++j) {
                float v = acc[mt][nt][j];
                int gm = bm + wm + mt * 16 + half * 8 + j;
                int gn = bn + wn + nt * 16 + lm;
                if (mode == 3) {
                    ((float*)Cout)[(size_t)gm * N + gn] = v;
                } else {
                    int b = gm >> 10, n = gm & 1023;
                    int vh = gn >> 6, dd = gn & 63;
                    int hh = vh >> 1, pp = vh & 1;
                    int vpos = n * 2 + pp;
                    bf16* o = (bf16*)Cout;
                    if (mode == 1)
                        o[((size_t)(b * 16 + hh) * 64 + dd) * SV + vpos] = (bf16)v;
                    else
                        o[((size_t)(b * 16 + hh) * SV + vpos) * 64 + dd] = (bf16)v;
                }
            }
}

// ---------------------------------------------------------------------------
// Causal flash attention over sv=2048, D=64, per (b,h).
// Grid: x = q-tile (64 rows), y = b*16+h. Block: 128 threads (4 waves),
// each wave owns 16 query rows and the full D=64 output.
__global__ void __launch_bounds__(128) flash_attn(
    const bf16* __restrict__ Q, const bf16* __restrict__ Kt,
    const bf16* __restrict__ V, bf16* __restrict__ Ov)
{
#if USE_ASYNC_LDS
    __shared__ __align__(16) bf16 Ks[3][64 * 72];   // K^T tile: d x keys
    __shared__ __align__(16) bf16 Vs[3][64 * 72];   // V tile: keys x d
#else
    __shared__ __align__(16) bf16 Ks[1][64 * 72];
    __shared__ __align__(16) bf16 Vs[1][64 * 72];
#endif
    __shared__ __align__(16) bf16 Ps[4][16 * 72];   // per-wave P staging

    const int tid = threadIdx.x;
    const int w = tid >> 5, lane = tid & 31;
    const int lm = lane & 15, half = lane >> 4;
    const int qt = blockIdx.x;          // 0..31
    const int bh = blockIdx.y;          // b*16 + h

    const bf16* Qb  = Q  + (size_t)bh * SV * 64;
    const bf16* Ktb = Kt + (size_t)bh * 64 * SV;
    const bf16* Vb  = V  + (size_t)bh * SV * 64;

    // Persistent Q fragments (16 rows x 64 d -> 2 A-frags)
    const int qrow = qt * 64 + w * 16 + lm;
    v16bf qf[2];
#pragma unroll
    for (int dc = 0; dc < 2; ++dc) {
        const bf16* p = Qb + (size_t)qrow * 64 + dc * 32 + half * 8;
        qf[dc] = cat8(*(const v8bf*)p, *(const v8bf*)(p + 16));
    }

    // cooperative K/V tile load indices: 32 bf16 (64B) per thread per array
    const int r = tid >> 1, cs = (tid & 1) * 32;

#if USE_ASYNC_LDS
    auto issue = [&](int kt, int buf) {
        const bf16* gk = Ktb + (size_t)r * SV + kt * 64 + cs;
        bf16* lk = &Ks[buf][r * 72 + cs];
        async_cp16(gk,      lk);
        async_cp16(gk + 8,  lk + 8);
        async_cp16(gk + 16, lk + 16);
        async_cp16(gk + 24, lk + 24);
        const bf16* gv = Vb + (size_t)(kt * 64 + r) * 64 + cs;
        bf16* lv = &Vs[buf][r * 72 + cs];
        async_cp16(gv,      lv);
        async_cp16(gv + 8,  lv + 8);
        async_cp16(gv + 16, lv + 16);
        async_cp16(gv + 24, lv + 24);
    };
    issue(0, 0);
#endif

    v8f of[4];
#pragma unroll
    for (int nf = 0; nf < 4; ++nf) of[nf] = vzero8();
    float mrow[8], lrow[8];
#pragma unroll
    for (int j = 0; j < 8; ++j) { mrow[j] = NEGBIG; lrow[j] = 0.0f; }

    for (int kt = 0; kt <= qt; ++kt) {
#if USE_ASYNC_LDS
        const int buf = kt % 3;
        if (kt < qt) {
            issue(kt + 1, (kt + 1) % 3);
            __builtin_amdgcn_s_wait_asynccnt(8);   // my tile-kt copies done
        } else {
            __builtin_amdgcn_s_wait_asynccnt(0);
        }
        __syncthreads();
#else
        const int buf = 0;
        __syncthreads();
        {
            const uint4* gk = (const uint4*)(Ktb + (size_t)r * SV + kt * 64 + cs);
            uint4* dk = (uint4*)&Ks[0][r * 72 + cs];
            dk[0] = gk[0]; dk[1] = gk[1]; dk[2] = gk[2]; dk[3] = gk[3];
            const uint4* gv = (const uint4*)(Vb + (size_t)(kt * 64 + r) * 64 + cs);
            uint4* dv = (uint4*)&Vs[0][r * 72 + cs];
            dv[0] = gv[0]; dv[1] = gv[1]; dv[2] = gv[2]; dv[3] = gv[3];
        }
        __syncthreads();
#endif

        // S = Q * K^T  (16 x 64 per wave)
        v8f sf[4];
#pragma unroll
        for (int nf = 0; nf < 4; ++nf) sf[nf] = vzero8();
#pragma unroll
        for (int dc = 0; dc < 2; ++dc) {
            const int kb = half * 8 + dc * 32;  // base d row in Ks (table order)
#pragma unroll
            for (int nf = 0; nf < 4; ++nf) {
                const bf16* p = &Ks[buf][kb * 72 + nf * 16 + lm];
                v16bf bb;
#pragma unroll
                for (int s = 0; s < 8; ++s) {
                    bb[s]     = p[s * 72];
                    bb[s + 8] = p[(s + 16) * 72];
                }
                sf[nf] = WMMA_BF16(qf[dc], bb, sf[nf]);
            }
        }

        // scale + causal mask (diagonal tile only)
        const bool diag = (kt == qt);
#pragma unroll
        for (int nf = 0; nf < 4; ++nf)
#pragma unroll
            for (int j = 0; j < 8; ++j) {
                float s = sf[nf][j] * SCALE;
                if (diag) {
                    int kv = nf * 16 + lm;
                    int qv = w * 16 + half * 8 + j;
                    s = (kv <= qv) ? s : NEGBIG;
                }
                sf[nf][j] = s;
            }

        // online softmax (row = 16 lanes within a half-wave)
#pragma unroll
        for (int j = 0; j < 8; ++j) {
            float rm = fmaxf(fmaxf(sf[0][j], sf[1][j]), fmaxf(sf[2][j], sf[3][j]));
            rm = fmaxf(rm, __shfl_xor(rm, 1, 32));
            rm = fmaxf(rm, __shfl_xor(rm, 2, 32));
            rm = fmaxf(rm, __shfl_xor(rm, 4, 32));
            rm = fmaxf(rm, __shfl_xor(rm, 8, 32));
            float mnew = fmaxf(mrow[j], rm);
            float corr = __expf(mrow[j] - mnew);
            mrow[j] = mnew;
            float rs = 0.0f;
#pragma unroll
            for (int nf = 0; nf < 4; ++nf) {
                float p = __expf(sf[nf][j] - mnew);
                sf[nf][j] = p;
                rs += p;
            }
            rs += __shfl_xor(rs, 1, 32);
            rs += __shfl_xor(rs, 2, 32);
            rs += __shfl_xor(rs, 4, 32);
            rs += __shfl_xor(rs, 8, 32);
            lrow[j] = lrow[j] * corr + rs;
#pragma unroll
            for (int nf = 0; nf < 4; ++nf) of[nf][j] *= corr;
        }

        // C-layout -> A-layout for P via per-wave LDS patch
#pragma unroll
        for (int nf = 0; nf < 4; ++nf)
#pragma unroll
            for (int j = 0; j < 8; ++j)
                Ps[w][(half * 8 + j) * 72 + nf * 16 + lm] = (bf16)sf[nf][j];

        // O += P * V
#pragma unroll
        for (int kc = 0; kc < 2; ++kc) {
            const bf16* pp = &Ps[w][lm * 72 + kc * 32 + half * 8];
            v16bf pa = cat8(*(const v8bf*)pp, *(const v8bf*)(pp + 16));
            const int vr = half * 8 + kc * 32;   // base key row in Vs (table order)
#pragma unroll
            for (int nf = 0; nf < 4; ++nf) {
                const bf16* vp = &Vs[buf][vr * 72 + nf * 16 + lm];
                v16bf vb;
#pragma unroll
                for (int s = 0; s < 8; ++s) {
                    vb[s]     = vp[s * 72];
                    vb[s + 8] = vp[(s + 16) * 72];
                }
                of[nf] = WMMA_BF16(pa, vb, of[nf]);
            }
        }
    }

    // Epilogue: normalize and scatter into virtual layout [b, n, (h*2+p)*64+d]
    const int b = bh >> 4, h = bh & 15;
#pragma unroll
    for (int j = 0; j < 8; ++j) {
        float inv = 1.0f / lrow[j];
        int vq = qt * 64 + w * 16 + half * 8 + j;
        int n = vq >> 1, p2 = vq & 1;
        size_t base = ((size_t)(b * 1024 + n)) * 2048 + (h * 2 + p2) * 64;
#pragma unroll
        for (int nf = 0; nf < 4; ++nf)
            Ov[base + nf * 16 + lm] = (bf16)(of[nf][j] * inv);
    }
}

// ---------------------------------------------------------------------------
extern "C" void kernel_launch(void* const* d_in, const int* in_sizes, int n_in,
                              void* d_out, int out_size, void* d_ws, size_t ws_size,
                              hipStream_t stream) {
    const float* x   = (const float*)d_in[0];
    const float* Wq  = (const float*)d_in[1];
    const float* Wk  = (const float*)d_in[2];
    const float* Wv  = (const float*)d_in[3];
    const float* Wo  = (const float*)d_in[4];
    const float* aq  = (const float*)d_in[5];
    const float* ak  = (const float*)d_in[6];
    const float* av  = (const float*)d_in[7];
    const float* col = (const float*)d_in[8];

    char* ws = (char*)d_ws;
    const size_t MB = 1u << 20;
    bf16* Xb  = (bf16*)(ws + 0 * MB);   // 2048 x 1024
    bf16* WqT = (bf16*)(ws + 4 * MB);   // 1024 x 2048 (K-major)
    bf16* WkT = (bf16*)(ws + 8 * MB);
    bf16* WvT = (bf16*)(ws + 12 * MB);
    bf16* WoT = (bf16*)(ws + 16 * MB);  // 2048 x 1024 (K-major)
    bf16* Qb  = (bf16*)(ws + 20 * MB);  // [b,h,vpos,d]
    bf16* Kb  = (bf16*)(ws + 28 * MB);  // [b,h,d,vpos]
    bf16* Vb  = (bf16*)(ws + 36 * MB);  // [b,h,vpos,d]
    bf16* Av  = (bf16*)(ws + 44 * MB);  // 2048 x 2048 virtual attn

    prep_x<<<2048, 256, 0, stream>>>(x, Xb);
    dim3 gw(4, 128);
    prep_weff<<<gw, 256, 0, stream>>>(Wq, aq, WqT);
    prep_weff<<<gw, 256, 0, stream>>>(Wk, ak, WkT);
    prep_weff<<<gw, 256, 0, stream>>>(Wv, av, WvT);
    prep_woeff<<<dim3(8, 64), 256, 0, stream>>>(Wo, col, WoT);

    gemm_wmma<<<dim3(16, 16), 256, 0, stream>>>(Xb, WqT, Qb, 2048, 2048, 1024, 0);
    gemm_wmma<<<dim3(16, 16), 256, 0, stream>>>(Xb, WkT, Kb, 2048, 2048, 1024, 1);
    gemm_wmma<<<dim3(16, 16), 256, 0, stream>>>(Xb, WvT, Vb, 2048, 2048, 1024, 2);

    flash_attn<<<dim3(32, 32), 128, 0, stream>>>(Qb, Kb, Vb, Av);

    gemm_wmma<<<dim3(8, 16), 256, 0, stream>>>(Av, WoT, d_out, 2048, 1024, 2048, 3);
}
